// TBRNN_76020921140335
// MI455X (gfx1250) — matrix operand order repaired and checked by reading, not compiled, not run on hardware
//
#include <hip/hip_runtime.h>
#include <math.h>

// ---------------- CDNA5 vector types ----------------
typedef __attribute__((ext_vector_type(16))) __bf16         v16bf;
typedef __attribute__((ext_vector_type(16))) unsigned short v16u;
typedef __attribute__((ext_vector_type(8)))  unsigned int   v8u;
typedef __attribute__((ext_vector_type(4)))  unsigned int   v4u;
typedef __attribute__((ext_vector_type(8)))  float          v8f;
typedef __attribute__((ext_vector_type(4)))  float          v4f;

#define BDIM    64
#define TDIM    512
#define INDIM   64
#define HIDDIM  256
#define KTOT    (HIDDIM * HIDDIM)   // 65536
#define SPLITS  64                  // split-K factor (K span 1024 per split)

__device__ __forceinline__ unsigned short f2bf(float f) {
  unsigned int u = __builtin_bit_cast(unsigned int, f);
  u += 0x7FFFu + ((u >> 16) & 1u);          // round-to-nearest-even
  return (unsigned short)(u >> 16);
}
__device__ __forceinline__ float bf2f(unsigned short h) {
  return __builtin_bit_cast(float, (unsigned int)h << 16);
}
// B fragment: 16 contiguous bf16 (32B) per lane, 32B aligned
__device__ __forceinline__ v16bf load_bfrag(const unsigned short* p) {
  v8u q = *(const v8u*)p;
  return __builtin_bit_cast(v16bf, q);
}
// packed bf16 multiply (VOP3P V_PK_MUL_BF16) — guaranteed native emission
__device__ __forceinline__ unsigned int pk_mul_bf16(unsigned int a, unsigned int b) {
  unsigned int d;
  asm("v_pk_mul_bf16 %0, %1, %2" : "=v"(d) : "v"(a), "v"(b));
  return d;
}
// A fragment: outer-product strip rj * r[k..], 16 bf16 via 8 packed muls
__device__ __forceinline__ v16bf make_afrag(unsigned int rjp, const v4u* rs) {
  const v4u lo = rs[0];
  const v4u hi = rs[1];   // +8 bf16 elements
  union { unsigned int u32[8]; v16u v; } af;
#pragma unroll
  for (int w = 0; w < 4; ++w) af.u32[w]     = pk_mul_bf16(rjp, lo[w]);
#pragma unroll
  for (int w = 0; w < 4; ++w) af.u32[4 + w] = pk_mul_bf16(rjp, hi[w]);
  return __builtin_bit_cast(v16bf, af.v);
}

// ---------------- prologue kernels ----------------
__global__ __launch_bounds__(256) void f32_to_bf16_kernel(
    const float* __restrict__ src, unsigned short* __restrict__ dst, int n) {
  int i = blockIdx.x * blockDim.x + threadIdx.x;
  int stride = gridDim.x * blockDim.x;
  for (; i < n; i += stride) dst[i] = f2bf(src[i]);
}

// input_I[t][b][i] = u[b][t][:] . W_in[i][:] + b_in[i]   (independent of scan)
__global__ __launch_bounds__(256) void input_proj_kernel(
    const float* __restrict__ u, const float* __restrict__ W_in,
    const float* __restrict__ b_in, float* __restrict__ inI) {
  __shared__ float us[INDIM];
  const int tb = blockIdx.x;          // tb = t*BDIM + b
  const int t = tb >> 6, b = tb & 63;
  if (threadIdx.x < INDIM)
    us[threadIdx.x] = u[((size_t)b * TDIM + t) * INDIM + threadIdx.x];
  __syncthreads();
  const int i = threadIdx.x;
  const float* w = W_in + (size_t)i * INDIM;
  float sum = b_in[i];
#pragma unroll 8
  for (int c = 0; c < INDIM; ++c) sum += us[c] * w[c];
  inI[(size_t)tb * HIDDIM + i] = sum;
}

__global__ __launch_bounds__(256) void init_state_kernel(
    const float* __restrict__ x0, float* __restrict__ x_cur,
    unsigned short* __restrict__ r_bf) {
  int i = blockIdx.x * 256 + threadIdx.x;
  float x = x0[i];
  x_cur[i] = x;
  r_bf[i] = f2bf(tanhf(x));
}

// ---------------- per-step GEMM: hidden = R2 @ Whh^T (split-K, bf16 WMMA) ---
// grid = (4 n-blocks of 64, SPLITS), block = 128 (4 waves).
// Wave w owns N-tile n0+w*16 (unique B fragment, W_hh read exactly once per
// step device-wide) and iterates all 4 batch tiles, rebuilding the cheap
// on-the-fly A fragment (r[b,j]*r[b,k]) with packed bf16 muls from LDS.
__global__ __launch_bounds__(128) void bilinear_gemm_kernel(
    const unsigned short* __restrict__ r_bf,    // [64][256] bf16
    const unsigned short* __restrict__ Whh_bf,  // [256][65536] bf16 (i, j*256+k)
    float* __restrict__ partials) {             // [SPLITS][64][256] f32
  __shared__ __align__(16) unsigned short r_s[BDIM * HIDDIM];  // 32 KB
  {
    const v4u* src = (const v4u*)r_bf;
    v4u* dst = (v4u*)r_s;
    for (int i = threadIdx.x; i < (BDIM * HIDDIM) / 8; i += 128) dst[i] = src[i];
  }
  __syncthreads();

  const int lane = threadIdx.x & 31;
  const int wave = threadIdx.x >> 5;
  const int half = lane >> 4;         // K-half lane group
  const int l16  = lane & 15;
  const int n0   = blockIdx.x * 64;
  const int s    = blockIdx.y;

  const int ncol = n0 + wave * 16 + l16;               // W_hh row (neuron i)
  const unsigned short* wrow = Whh_bf + (size_t)ncol * KTOT;

  // LDS row bases for this lane's 4 batch rows (one per M-tile)
  const unsigned short* r0 = r_s + (size_t)(0 * 16 + l16) * HIDDIM;
  const unsigned short* r1 = r_s + (size_t)(1 * 16 + l16) * HIDDIM;
  const unsigned short* r2 = r_s + (size_t)(2 * 16 + l16) * HIDDIM;
  const unsigned short* r3 = r_s + (size_t)(3 * 16 + l16) * HIDDIM;

  v8f acc0 = {}, acc1 = {}, acc2 = {}, acc3 = {};

  for (int jb = 0; jb < 4; ++jb) {
    const int j = s * 4 + jb;                          // outer index of R2
    // splat r[b,j] into both bf16 halves, one per M-tile
    const unsigned int q0 = (unsigned int)r0[j] * 0x00010001u;
    const unsigned int q1 = (unsigned int)r1[j] * 0x00010001u;
    const unsigned int q2 = (unsigned int)r2[j] * 0x00010001u;
    const unsigned int q3 = (unsigned int)r3[j] * 0x00010001u;
    for (int c8 = 0; c8 < 8; ++c8) {
      const int kba = c8 * 32 + half * 16;             // lane's A K-base (k)
      // shared B fragment: 32 contiguous bf16 of W_hh row
      const v16bf b = load_bfrag(wrow + (size_t)j * 256 + c8 * 32 + half * 16);
      const v16bf a0 = make_afrag(q0, (const v4u*)(r0 + kba));
      const v16bf a1 = make_afrag(q1, (const v4u*)(r1 + kba));
      const v16bf a2 = make_afrag(q2, (const v4u*)(r2 + kba));
      const v16bf a3 = make_afrag(q3, (const v4u*)(r3 + kba));
      acc0 = __builtin_amdgcn_wmma_f32_16x16x32_bf16(false, a0, false, b,
                                                     (short)0, acc0, false, false);
      acc1 = __builtin_amdgcn_wmma_f32_16x16x32_bf16(false, a1, false, b,
                                                     (short)0, acc1, false, false);
      acc2 = __builtin_amdgcn_wmma_f32_16x16x32_bf16(false, a2, false, b,
                                                     (short)0, acc2, false, false);
      acc3 = __builtin_amdgcn_wmma_f32_16x16x32_bf16(false, a3, false, b,
                                                     (short)0, acc3, false, false);
    }
  }

  // ---- write split-K partials (f32 C/D layout: lane<16 -> M=v, else M=v+8) ----
  float* pr = partials + (size_t)s * (BDIM * HIDDIM);
  const int col = n0 + wave * 16 + l16;
#pragma unroll
  for (int v = 0; v < 8; ++v) {
    const int r = v + (half ? 8 : 0);
    pr[(size_t)(0 * 16 + r) * HIDDIM + col] = acc0[v];
    pr[(size_t)(1 * 16 + r) * HIDDIM + col] = acc1[v];
    pr[(size_t)(2 * 16 + r) * HIDDIM + col] = acc2[v];
    pr[(size_t)(3 * 16 + r) * HIDDIM + col] = acc3[v];
  }
}

// ---------------- per-step state update ----------------
__global__ __launch_bounds__(256) void state_update_kernel(
    const float* __restrict__ partials, const float* __restrict__ noise,
    const float* __restrict__ inI, float* __restrict__ x_cur,
    unsigned short* __restrict__ r_bf, float* __restrict__ traj,
    float* __restrict__ x_final, int t) {
  const int idx = blockIdx.x * 256 + threadIdx.x;   // b*256+i
  float h = 0.f;
#pragma unroll 8
  for (int s = 0; s < SPLITS; ++s) h += partials[(size_t)s * (BDIM * HIDDIM) + idx];
  const float x = x_cur[idx];
  const float xn = x + 0.05f * noise[(size_t)t * (BDIM * HIDDIM) + idx]
                     + 0.2f * (-x + h + inI[(size_t)t * (BDIM * HIDDIM) + idx]);
  x_cur[idx] = xn;
  r_bf[idx] = f2bf(tanhf(xn));
  const int b = idx >> 8, i = idx & 255;
  traj[((size_t)b * TDIM + t) * HIDDIM + i] = xn;   // [B,T,HID]
  if (t == TDIM - 1) x_final[idx] = xn;
}

// ---------------- epilogue: output = tanh(traj) @ W_out^T + b_out (WMMA) ----
// M = B*T = 32768 rows, N = 64, K = 256. grid = 512, block = 128 (4 waves).
__global__ __launch_bounds__(128) void outproj_kernel(
    const float* __restrict__ traj, const unsigned short* __restrict__ Wout_bf,
    const float* __restrict__ b_out, float* __restrict__ out) {
  const int lane = threadIdx.x & 31;
  const int wave = threadIdx.x >> 5;
  const int half = lane >> 4;
  const int l16  = lane & 15;
  const int m0   = blockIdx.x * 64 + wave * 16;
  const int mrow = m0 + l16;

  v8f acc0 = {}, acc1 = {}, acc2 = {}, acc3 = {};
  for (int c = 0; c < HIDDIM / 32; ++c) {
    const int kb = c * 32;
    // A fragment: tanh of trajectory row, converted to bf16 on the fly
    const v4f* tp = (const v4f*)(traj + (size_t)mrow * HIDDIM + kb + (half ? 16 : 0));
    const v4f f0 = tp[0], f1 = tp[1], f2 = tp[2], f3 = tp[3];
    union { unsigned short u16[16]; v16u v; } af;
#pragma unroll
    for (int e = 0; e < 4; ++e) {
      af.u16[e]      = f2bf(tanhf(f0[e]));
      af.u16[4 + e]  = f2bf(tanhf(f1[e]));
      af.u16[8 + e]  = f2bf(tanhf(f2[e]));
      af.u16[12 + e] = f2bf(tanhf(f3[e]));
    }
    const v16bf a = __builtin_bit_cast(v16bf, af.v);
    const size_t bcol = (size_t)kb + (half ? 16 : 0);
    const unsigned short* wb = Wout_bf + (size_t)l16 * HIDDIM + bcol;
    acc0 = __builtin_amdgcn_wmma_f32_16x16x32_bf16(false, a, false,
              load_bfrag(wb + (size_t)0 * 16 * HIDDIM), (short)0, acc0, false, false);
    acc1 = __builtin_amdgcn_wmma_f32_16x16x32_bf16(false, a, false,
              load_bfrag(wb + (size_t)1 * 16 * HIDDIM), (short)0, acc1, false, false);
    acc2 = __builtin_amdgcn_wmma_f32_16x16x32_bf16(false, a, false,
              load_bfrag(wb + (size_t)2 * 16 * HIDDIM), (short)0, acc2, false, false);
    acc3 = __builtin_amdgcn_wmma_f32_16x16x32_bf16(false, a, false,
              load_bfrag(wb + (size_t)3 * 16 * HIDDIM), (short)0, acc3, false, false);
  }
  const float bo0 = b_out[l16], bo1 = b_out[16 + l16];
  const float bo2 = b_out[32 + l16], bo3 = b_out[48 + l16];
#pragma unroll
  for (int v = 0; v < 8; ++v) {
    const int row = m0 + v + (half ? 8 : 0);
    float* orow = out + (size_t)row * INDIM + l16;
    orow[0]  = acc0[v] + bo0;
    orow[16] = acc1[v] + bo1;
    orow[32] = acc2[v] + bo2;
    orow[48] = acc3[v] + bo3;
  }
}

// ---------------- launch ----------------
extern "C" void kernel_launch(void* const* d_in, const int* in_sizes, int n_in,
                              void* d_out, int out_size, void* d_ws, size_t ws_size,
                              hipStream_t stream) {
  const float* u     = (const float*)d_in[0];
  const float* x0    = (const float*)d_in[1];
  const float* noise = (const float*)d_in[2];
  const float* W_hh  = (const float*)d_in[3];
  const float* W_in  = (const float*)d_in[4];
  const float* b_in  = (const float*)d_in[5];
  const float* W_out = (const float*)d_in[6];
  const float* b_out = (const float*)d_in[7];

  char* ws = (char*)d_ws;   // layout (bytes), ~72 MB total
  unsigned short* Whh_bf  = (unsigned short*)(ws + 0);          // 33,554,432
  float*          inI     = (float*)(ws + 33554432);            // 33,554,432
  float*          partials= (float*)(ws + 67108864);            //  4,194,304
  float*          x_cur   = (float*)(ws + 71303168);            //     65,536
  unsigned short* r_bf    = (unsigned short*)(ws + 71368704);   //     32,768
  unsigned short* Wout_bf = (unsigned short*)(ws + 71401472);   //     32,768

  float* out     = (float*)d_out;                   // [B,T,IN]
  float* x_final = out + (size_t)BDIM * TDIM * INDIM;           // +2,097,152
  float* traj    = x_final + (size_t)BDIM * HIDDIM;             // +16,384

  // prologue
  f32_to_bf16_kernel<<<2048, 256, 0, stream>>>(W_hh, Whh_bf, KTOT * HIDDIM);
  f32_to_bf16_kernel<<<16, 256, 0, stream>>>(W_out, Wout_bf, INDIM * HIDDIM);
  input_proj_kernel<<<TDIM * BDIM, HIDDIM, 0, stream>>>(u, W_in, b_in, inI);
  init_state_kernel<<<BDIM, HIDDIM, 0, stream>>>(x0, x_cur, r_bf);

  // sequential scan: per step, one device-filling split-K WMMA GEMM + update
  for (int t = 0; t < TDIM; ++t) {
    bilinear_gemm_kernel<<<dim3(4, SPLITS), 128, 0, stream>>>(r_bf, Whh_bf, partials);
    state_update_kernel<<<BDIM, HIDDIM, 0, stream>>>(partials, noise, inI, x_cur,
                                                     r_bf, traj, x_final, t);
  }

  // epilogue: output projection
  outproj_kernel<<<(BDIM * TDIM) / 64, 128, 0, stream>>>(traj, Wout_bf, b_out, out);
}
